// ArsenalAttention_14396730376712
// MI455X (gfx1250) — compile-verified, matrix operand order checked by reading, not run on hardware
//
#include <hip/hip_runtime.h>

typedef __attribute__((ext_vector_type(16))) __bf16 v16bf;
typedef __attribute__((ext_vector_type(8)))  __bf16 v8bf;
typedef __attribute__((ext_vector_type(8)))  float  v8f;

#define HIDDEN  2048
#define NHEADS  16
#define HEADDIM 128
#define BATCH   2
#define SEQ     2048
#define BSROWS  (BATCH * SEQ)   /* 4096 */

#define TM 128
#define TN 128
#define TK 32
#define LDSS 40   /* TK + 8 pad: 80B rows keep 16B alignment and stagger banks */

__device__ __forceinline__ unsigned short f32_to_bf16_rne(float f) {
  unsigned int u = __builtin_bit_cast(unsigned int, f);
  unsigned int r = u + 0x7FFFu + ((u >> 16) & 1u);
  return (unsigned short)(r >> 16);
}

// CDNA5 async global->LDS copy (16 bytes per lane), tracked by ASYNCcnt.
// VDST VGPR carries the LDS byte address (low 32 bits of the flat shared ptr).
__device__ __forceinline__ void async_copy_b128(const void* gsrc, void* lds_dst) {
  unsigned lds_off = (unsigned)(unsigned long long)(uintptr_t)lds_dst;
  asm volatile("global_load_async_to_lds_b128 %0, %1, off"
               :: "v"(lds_off), "v"(gsrc) : "memory");
}
__device__ __forceinline__ void wait_asynccnt0() {
  asm volatile("s_wait_asynccnt 0x0" ::: "memory");
}

// ---------------------------------------------------------------------------
// elementwise f32 -> bf16 conversion
// ---------------------------------------------------------------------------
__global__ __launch_bounds__(256) void aa_cvt_f32_bf16(const float* __restrict__ in,
                                                       unsigned short* __restrict__ out,
                                                       int n) {
  for (int i = blockIdx.x * 256 + threadIdx.x; i < n; i += gridDim.x * 256)
    out[i] = f32_to_bf16_rne(in[i]);
}

// ---------------------------------------------------------------------------
// Generic batched bf16 WMMA GEMM:  C[z] = A[z] (MxK) * B[z] (KxN)
//   flags bit0: B is stored as [N][K] (row n of B-source = column n of B)
//   flags bit1: A is f32 in memory (convert to bf16 while staging to LDS)
//   flags bit2: write C as bf16 (else f32)
//   flags bit3: causal epilogue: n>m -> -inf, else *scale (C must be f32)
// Block tile 128x128, 8 waves (wave32), each wave 32x64 (2x4 WMMA 16x16 tiles).
// Double-buffered LDS; bf16 tiles staged with async global->LDS copies.
// ---------------------------------------------------------------------------
__global__ __launch_bounds__(256) void aa_gemm_wmma_bf16(
    const void* __restrict__ Av, const unsigned short* __restrict__ Bp,
    void* __restrict__ Cv,
    int lda, int ldb, int ldc, int K, int Hdim,
    long long sAb, long long sAh, long long sBb, long long sBh,
    long long sCb, long long sCh,
    int flags, float scale)
{
  const bool b_nk   = (flags & 1) != 0;
  const bool a_f32  = (flags & 2) != 0;
  const bool out_bf = (flags & 4) != 0;
  const bool causal = (flags & 8) != 0;

  __shared__ unsigned short As[2][TM][LDSS];
  __shared__ unsigned short Bt[2][TN][LDSS];   // B staged N-major: Bt[buf][n][k]

  const int z  = blockIdx.z;
  const int bb = z / Hdim, hh = z % Hdim;
  const long long aOff = (long long)bb * sAb + (long long)hh * sAh;
  const long long bOff = (long long)bb * sBb + (long long)hh * sBh;
  const long long cOff = (long long)bb * sCb + (long long)hh * sCh;

  const int rowBase = blockIdx.y * TM;
  const int colBase = blockIdx.x * TN;
  const int tid = threadIdx.x;

  // Fully masked tile: skip the K loop entirely, just emit -inf.
  if (causal && colBase > rowBase + (TM - 1)) {
    float* C = (float*)Cv + cOff;
    for (int i = tid; i < TM * TN; i += 256) {
      const int m = i >> 7, n = i & 127;
      C[(long long)(rowBase + m) * ldc + (colBase + n)] = -__builtin_inff();
    }
    return;
  }

  const int wave = tid >> 5, lane = tid & 31;
  const int wm = (wave >> 1) * 32;   // 4 wave-rows * 32
  const int wn = (wave & 1) * 64;    // 2 wave-cols * 64
  const int fr = lane & 15;          // row (A) / col (B) within 16x16 tile
  const int hi = lane >> 4;          // half-wave select
  const int kb = hi * 8;             // lanes 0-15 -> K{0..7,16..23}, 16-31 -> K{8..15,24..31}

  v8f acc[2][4];
  #pragma unroll
  for (int mi = 0; mi < 2; ++mi)
    #pragma unroll
    for (int ni = 0; ni < 4; ++ni)
      #pragma unroll
      for (int j = 0; j < 8; ++j) acc[mi][ni][j] = 0.0f;

  const unsigned short* Abf = (const unsigned short*)Av;
  const float*          Af  = (const float*)Av;

  // Stage one K-step tile (A and B) into LDS buffer `buf`.
  auto stage = [&](int buf, int k0) {
    {
      const int r = tid >> 1;
      const int c = (tid & 1) * 16;
      const long long g = aOff + (long long)(rowBase + r) * lda + (k0 + c);
      if (a_f32) {
        #pragma unroll
        for (int i = 0; i < 16; ++i)
          As[buf][r][c + i] = f32_to_bf16_rne(Af[g + i]);
      } else {
        async_copy_b128(Abf + g,     &As[buf][r][c]);
        async_copy_b128(Abf + g + 8, &As[buf][r][c + 8]);
      }
    }
    if (b_nk) {
      const int r = tid >> 1;
      const int c = (tid & 1) * 16;
      const long long g = bOff + (long long)(colBase + r) * ldb + (k0 + c);
      async_copy_b128(Bp + g,     &Bt[buf][r][c]);
      async_copy_b128(Bp + g + 8, &Bt[buf][r][c + 8]);
    } else {
      const int kk = tid >> 3;            // 0..31
      const int ns = (tid & 7) * 16;      // 0,16,...,112
      const long long g = bOff + (long long)(k0 + kk) * ldb + (colBase + ns);
      #pragma unroll
      for (int i = 0; i < 16; ++i)
        Bt[buf][ns + i][kk] = Bp[g + i];
    }
  };

  const int nsteps = K / TK;
  stage(0, 0);
  wait_asynccnt0();
  __syncthreads();

  for (int s = 0; s < nsteps; ++s) {
    const int cur = s & 1;
    if (s + 1 < nsteps) stage(cur ^ 1, (s + 1) * TK);   // prefetch next tile

    // ---- build fragments (two ds_load_b128 per 16x16 tile) ----
    v16bf afrag[2], bfrag[4];
    #pragma unroll
    for (int mi = 0; mi < 2; ++mi) {
      const v8bf lo  = *(const v8bf*)&As[cur][wm + mi * 16 + fr][kb];
      const v8bf hi8 = *(const v8bf*)&As[cur][wm + mi * 16 + fr][kb + 16];
      #pragma unroll
      for (int i = 0; i < 8; ++i) { afrag[mi][i] = lo[i]; afrag[mi][8 + i] = hi8[i]; }
    }
    #pragma unroll
    for (int ni = 0; ni < 4; ++ni) {
      const v8bf lo  = *(const v8bf*)&Bt[cur][wn + ni * 16 + fr][kb];
      const v8bf hi8 = *(const v8bf*)&Bt[cur][wn + ni * 16 + fr][kb + 16];
      #pragma unroll
      for (int i = 0; i < 8; ++i) { bfrag[ni][i] = lo[i]; bfrag[ni][8 + i] = hi8[i]; }
    }

    // ---- 8 WMMAs per wave per K-step ----
    #pragma unroll
    for (int mi = 0; mi < 2; ++mi)
      #pragma unroll
      for (int ni = 0; ni < 4; ++ni)
        acc[mi][ni] = __builtin_amdgcn_wmma_f32_16x16x32_bf16(
            false, afrag[mi], false, bfrag[ni], (short)0, acc[mi][ni], false, false);

    wait_asynccnt0();   // our async writes into the other buffer are done
    __syncthreads();    // everyone's staging (async + ds) visible
  }

  // ---- epilogue ----
  #pragma unroll
  for (int mi = 0; mi < 2; ++mi)
    #pragma unroll
    for (int ni = 0; ni < 4; ++ni)
      #pragma unroll
      for (int j = 0; j < 8; ++j) {
        const int gm = rowBase + wm + mi * 16 + hi * 8 + j;
        const int gn = colBase + wn + ni * 16 + fr;
        float v = acc[mi][ni][j];
        if (causal) v = (gn > gm) ? -__builtin_inff() : v * scale;
        const long long idx = cOff + (long long)gm * ldc + gn;
        if (out_bf) ((unsigned short*)Cv)[idx] = f32_to_bf16_rne(v);
        else        ((float*)Cv)[idx] = v;
      }
}

// ---------------------------------------------------------------------------
// In-place row softmax over ncols f32 (one block per row)
// ---------------------------------------------------------------------------
__global__ __launch_bounds__(256) void aa_softmax_rows(float* __restrict__ P, int ncols) {
  __shared__ float red[256];
  float* p = P + (long long)blockIdx.x * ncols;
  const int t = threadIdx.x;

  float m = -__builtin_inff();
  for (int c = t; c < ncols; c += 256) m = fmaxf(m, p[c]);
  red[t] = m; __syncthreads();
  for (int s = 128; s > 0; s >>= 1) {
    if (t < s) red[t] = fmaxf(red[t], red[t + s]);
    __syncthreads();
  }
  const float rowmax = red[0];
  __syncthreads();

  float sum = 0.0f;
  for (int c = t; c < ncols; c += 256) {
    const float e = __expf(p[c] - rowmax);   // exp(-inf) = 0 handles the mask
    p[c] = e;
    sum += e;
  }
  red[t] = sum; __syncthreads();
  for (int s = 128; s > 0; s >>= 1) {
    if (t < s) red[t] += red[t + s];
    __syncthreads();
  }
  const float inv = 1.0f / red[0];
  __syncthreads();
  for (int c = t; c < ncols; c += 256) p[c] *= inv;
}

// ---------------------------------------------------------------------------
extern "C" void kernel_launch(void* const* d_in, const int* in_sizes, int n_in,
                              void* d_out, int out_size, void* d_ws, size_t ws_size,
                              hipStream_t stream) {
  const float* X  = (const float*)d_in[0];
  const float* Wq = (const float*)d_in[1];
  const float* Wk = (const float*)d_in[2];
  const float* Wv = (const float*)d_in[3];
  const float* Wo = (const float*)d_in[4];

  float* out  = (float*)d_out;                              // [B,S,2048]
  float* attn = out + (long long)BSROWS * HIDDEN;           // [B,H,S,S]

  // workspace layout (bf16 stored as ushort): ~112 MB total
  unsigned short* Xbf  = (unsigned short*)d_ws;
  unsigned short* Wqbf = Xbf  + (long long)BSROWS * HIDDEN;
  unsigned short* Wkbf = Wqbf + (long long)HIDDEN * HIDDEN;
  unsigned short* Wvbf = Wkbf + (long long)HIDDEN * HIDDEN;
  unsigned short* Wobf = Wvbf + (long long)HIDDEN * HIDDEN;
  unsigned short* Qbf  = Wobf + (long long)HIDDEN * HIDDEN;
  unsigned short* Kbf  = Qbf  + (long long)BSROWS * HIDDEN;
  unsigned short* Vbf  = Kbf  + (long long)BSROWS * HIDDEN;
  unsigned short* Ctx  = Vbf  + (long long)BSROWS * HIDDEN;

  const long long SH = (long long)SEQ * HIDDEN;             // per-batch stride of [S,2048]
  const long long SS = (long long)SEQ * SEQ;

  dim3 blk(256);

  // 1) f32 -> bf16 conversions
  aa_cvt_f32_bf16<<<4096, blk, 0, stream>>>(X,  Xbf,  BSROWS * HIDDEN);
  aa_cvt_f32_bf16<<<2048, blk, 0, stream>>>(Wq, Wqbf, HIDDEN * HIDDEN);
  aa_cvt_f32_bf16<<<2048, blk, 0, stream>>>(Wk, Wkbf, HIDDEN * HIDDEN);
  aa_cvt_f32_bf16<<<2048, blk, 0, stream>>>(Wv, Wvbf, HIDDEN * HIDDEN);
  aa_cvt_f32_bf16<<<2048, blk, 0, stream>>>(Wo, Wobf, HIDDEN * HIDDEN);

  // 2) Q/K/V projections: [4096x2048] = Xbf @ Wbf   (out bf16)
  {
    dim3 g(HIDDEN / TN, BSROWS / TM, 1);
    aa_gemm_wmma_bf16<<<g, blk, 0, stream>>>(Xbf, Wqbf, Qbf, HIDDEN, HIDDEN, HIDDEN,
        HIDDEN, 1, 0, 0, 0, 0, 0, 0, /*out_bf*/4, 1.0f);
    aa_gemm_wmma_bf16<<<g, blk, 0, stream>>>(Xbf, Wkbf, Kbf, HIDDEN, HIDDEN, HIDDEN,
        HIDDEN, 1, 0, 0, 0, 0, 0, 0, 4, 1.0f);
    aa_gemm_wmma_bf16<<<g, blk, 0, stream>>>(Xbf, Wvbf, Vbf, HIDDEN, HIDDEN, HIDDEN,
        HIDDEN, 1, 0, 0, 0, 0, 0, 0, 4, 1.0f);
  }

  // 3) scores[b,h] = Q[b,:,h*128:] @ K[b,:,h*128:]^T ; mask + /sqrt(128) in epilogue
  {
    dim3 g(SEQ / TN, SEQ / TM, BATCH * NHEADS);
    aa_gemm_wmma_bf16<<<g, blk, 0, stream>>>(Qbf, Kbf, attn,
        HIDDEN, HIDDEN, SEQ, HEADDIM, NHEADS,
        SH, HEADDIM,                 /* A strides (b,h) */
        SH, HEADDIM,                 /* B strides (b,h) — K^T is K's [N][K] layout */
        (long long)NHEADS * SS, SS,  /* C strides (b,h) */
        /*b_nk|causal*/ 1 | 8, 0.08838834764831845f);
  }

  // 4) softmax rows of attn (in place, f32 probabilities stay in d_out)
  aa_softmax_rows<<<BATCH * NHEADS * SEQ, blk, 0, stream>>>(attn, SEQ);

  // 5) context[b,:,h*128:] = P[b,h] (f32, converted on the fly) @ V[b,:,h*128:]
  {
    dim3 g(HEADDIM / TN, SEQ / TM, BATCH * NHEADS);   // 1 x 16 x 32
    aa_gemm_wmma_bf16<<<g, blk, 0, stream>>>(attn, Vbf, Ctx,
        SEQ, HIDDEN, HIDDEN, SEQ, NHEADS,
        (long long)NHEADS * SS, SS,  /* A = probabilities */
        SH, HEADDIM,                 /* B = V */
        SH, HEADDIM,                 /* C = context columns h*128.. */
        /*a_f32|out_bf*/ 2 | 4, 1.0f);
  }

  // 6) out = context @ Wo  (f32 output)
  {
    dim3 g(HIDDEN / TN, BSROWS / TM, 1);
    aa_gemm_wmma_bf16<<<g, blk, 0, stream>>>(Ctx, Wobf, out, HIDDEN, HIDDEN, HIDDEN,
        HIDDEN, 1, 0, 0, 0, 0, 0, 0, /*f32 out*/0, 1.0f);
  }
}
